// MSAColGlobalAttention_55851754717709
// MI455X (gfx1250) — compile-verified, matrix operand order checked by reading, not run on hardware
//
#include <hip/hip_runtime.h>
#include <hip/hip_fp16.h>

// ---------------- problem constants (fixed by the reference) ----------------
#define N_SEQ 512
#define L_SEQ 1024
#define D_MSA 64
#define HD    64          // H*DH
#define NPOS  (N_SEQ * L_SEQ)

// ---------------- workspace layout (bytes) ----------------
// xbf16 : NPOS*64 f16      = 67,108,864
// k_ws  : NPOS*8  f32      = 16,777,216
// v_ws  : NPOS*8  f32      = 16,777,216
// q_acc : L*64 f32         =    262,144
// o_ws  : L*64 f32         =    262,144
// wpack : 14336 f16        =     28,672
#define OFF_XBF 0
#define OFF_K   67108864ull
#define OFF_V   83886080ull
#define OFF_Q   100663296ull
#define OFF_O   100925440ull
#define OFF_WP  101187584ull

// wpack element offsets (u16 units)
#define WP_Q   0        // 4 tiles  = 4096
#define WP_K   4096     // 1 tile   = 1024
#define WP_V   5120     // 1 tile   = 1024
#define WP_G   6144     // 4 tiles  = 4096
#define WP_O   10240    // 4 tiles  = 4096
#define WP_QKV_U16 6144 // Wq+Wk+Wv span
#define WP_GO_U16  8192 // Wg+Wo span

typedef __attribute__((ext_vector_type(16))) _Float16 v16h;
typedef __attribute__((ext_vector_type(8)))  float    v8f;

struct alignas(16) U128 { unsigned int w[4]; };

union AFrag { v16h v; U128 q[2]; unsigned short s[16]; };

static __device__ inline unsigned short f2h_bits(float f) {
    _Float16 h = (_Float16)f;
    unsigned short u;
    __builtin_memcpy(&u, &h, 2);
    return u;
}

// A-matrix fragment (16x32 f16) from a row-major [16][64] f16 tile, per ISA
// layout: lane<16 holds row=lane, K = {kbase..kbase+7} U {kbase+16..kbase+23},
// kbase = (lane>>4)*8, all offset by kk*32. Each chunk is 16 contiguous bytes.
static __device__ inline v16h load_a_frag(const unsigned short* tile, int lane, int kk) {
    const int row   = lane & 15;
    const int kbase = ((lane >> 4) << 3) + (kk << 5);
    AFrag f;
    f.q[0] = *(const U128*)(tile + row * 64 + kbase);
    f.q[1] = *(const U128*)(tile + row * 64 + kbase + 16);
    return f.v;
}

// B-matrix fragment: frag (t,kk) for this lane lives at 32 contiguous bytes.
static __device__ inline v16h load_b_frag(const unsigned short* wp, int t, int kk, int lane) {
    AFrag f;
    const U128* p = (const U128*)(wp + (size_t)t * 1024 + (size_t)kk * 512 + (size_t)lane * 16);
    f.q[0] = p[0];
    f.q[1] = p[1];
    return f.v;
}

#define WMMA_F16(A, B, C) \
    __builtin_amdgcn_wmma_f32_16x16x32_f16(false, (A), false, (B), (short)0, (C), false, false)

// ---------------------------------------------------------------------------
// K0: pack all five weight matrices (f32, row-major [64][cols]) into f16
// B-fragment order: [tile t][kk][lane][elem e], elem e -> K = kk*32 + kbase +
// e + (e>=8 ? 8 : 0), col = t*16 + (lane&15). Padded cols (Wk/Wv) -> 0.
// ---------------------------------------------------------------------------
__global__ void __launch_bounds__(256)
prepack_weights(const float* __restrict__ Wq, const float* __restrict__ Wk,
                const float* __restrict__ Wv, const float* __restrict__ Wg,
                const float* __restrict__ Wo, unsigned short* __restrict__ wpack) {
    const float* srcs[5]   = {Wq, Wk, Wv, Wg, Wo};
    const int    colsA[5]  = {64, 8, 8, 64, 64};
    const int    tilesA[5] = {4, 1, 1, 4, 4};
    unsigned short* dst = wpack;
    for (int m = 0; m < 5; ++m) {
        const int total = tilesA[m] * 2 * 32 * 16;
        for (int i = threadIdx.x; i < total; i += blockDim.x) {
            const int e    = i & 15;
            const int lane = (i >> 4) & 31;
            const int kk   = (i >> 9) & 1;
            const int t    = i >> 10;
            const int kbase = ((lane >> 4) << 3);
            const int K   = kk * 32 + kbase + e + ((e >= 8) ? 8 : 0);
            const int col = t * 16 + (lane & 15);
            const float val = (col < colsA[m]) ? srcs[m][(size_t)K * colsA[m] + col] : 0.0f;
            dst[i] = f2h_bits(val);
        }
        dst += total;
    }
}

// ---------------------------------------------------------------------------
// K1: LayerNorm + Q/K/V projections.
// Grid 4096, block 256 (8 waves). Block covers one 16-row l-tile and 8 n's
// (one per wave). Weights staged to LDS once per block (copies overlap LN).
// q partials reduced over the block's 8 n via LDS atomics, then one global
// atomicAdd pass per block.
// ---------------------------------------------------------------------------
__global__ void __launch_bounds__(256)
ln_qkv_kernel(const float* __restrict__ msa, const float* __restrict__ ln_w,
              const float* __restrict__ ln_b, unsigned short* __restrict__ xbf,
              float* __restrict__ k_ws, float* __restrict__ v_ws,
              float* __restrict__ q_acc, const unsigned short* __restrict__ wpack) {
    __shared__ __attribute__((aligned(16))) unsigned short tiles[8][16 * 64];
    __shared__ __attribute__((aligned(16))) unsigned short wlds[WP_QKV_U16];
    __shared__ float qred[16][64];

    const int tid  = threadIdx.x;
    const int wave = tid >> 5;
    const int lane = tid & 31;
    const int lt = blockIdx.x & 63;         // l-tile index
    const int ng = blockIdx.x >> 6;         // n-group index
    const int n  = ng * 8 + wave;
    const int l0 = lt * 16;
    const int p0 = n * L_SEQ + l0;

    // stage Wq|Wk|Wv fragments into LDS (768 x 16B), overlaps the LN below
    {
        const U128* src = (const U128*)(wpack + WP_Q);
        U128* dst = (U128*)wlds;
        for (int i = tid; i < WP_QKV_U16 / 8; i += 256) dst[i] = src[i];
    }
    // zero the q reduction buffer (covered by the barrier below)
    for (int i = tid; i < 16 * 64; i += 256) ((float*)qred)[i] = 0.0f;

    // ---- LayerNorm: 2 lanes per row, 32 f32 each ----
    const int row  = lane >> 1;
    const int half = lane & 1;
    const float* src = msa + (size_t)(p0 + row) * 64 + half * 32;
    float vals[32];
    float s = 0.0f, ss = 0.0f;
#pragma unroll
    for (int i = 0; i < 8; ++i) {
        float4 f = ((const float4*)src)[i];
        vals[i * 4 + 0] = f.x; vals[i * 4 + 1] = f.y;
        vals[i * 4 + 2] = f.z; vals[i * 4 + 3] = f.w;
        s  += f.x + f.y + f.z + f.w;
        ss += f.x * f.x + f.y * f.y + f.z * f.z + f.w * f.w;
    }
    s  += __shfl_xor(s, 1, 32);
    ss += __shfl_xor(ss, 1, 32);
    const float mean = s * (1.0f / 64.0f);
    const float var  = ss * (1.0f / 64.0f) - mean * mean;
    const float rs   = rsqrtf(var + 1e-5f);

    __attribute__((aligned(16))) unsigned short outb[32];
#pragma unroll
    for (int i = 0; i < 32; ++i) {
        const int d = half * 32 + i;
        outb[i] = f2h_bits((vals[i] - mean) * rs * ln_w[d] + ln_b[d]);
    }
    {
        U128* lds4 = (U128*)&tiles[wave][row * 64 + half * 32];
        U128* g4   = (U128*)(xbf + (size_t)(p0 + row) * 64 + half * 32);
        const U128* o4 = (const U128*)outb;
#pragma unroll
        for (int i = 0; i < 4; ++i) { lds4[i] = o4[i]; g4[i] = o4[i]; }
    }
    __syncthreads();

    // ---- WMMA projections (B fragments now from LDS) ----
    const v16h a0 = load_a_frag(tiles[wave], lane, 0);
    const v16h a1 = load_a_frag(tiles[wave], lane, 1);
    const int mbase = (lane >> 4) << 3;
    const int coll  = lane & 15;
    const unsigned short* wq = wlds;                    // 4 tiles
    const unsigned short* wk = wlds + WP_K;             // 1 tile (padded)
    const unsigned short* wv = wlds + WP_V;             // 1 tile (padded)

#pragma unroll
    for (int t = 0; t < 4; ++t) {              // q -> LDS reduce over n
        v8f acc = {};
        acc = WMMA_F16(a0, load_b_frag(wq, t, 0, lane), acc);
        acc = WMMA_F16(a1, load_b_frag(wq, t, 1, lane), acc);
        const int col = t * 16 + coll;
#pragma unroll
        for (int r = 0; r < 8; ++r) atomicAdd(&qred[mbase + r][col], acc[r]);
    }
    {                                           // k
        v8f acc = {};
        acc = WMMA_F16(a0, load_b_frag(wk, 0, 0, lane), acc);
        acc = WMMA_F16(a1, load_b_frag(wk, 0, 1, lane), acc);
        if (coll < 8) {
#pragma unroll
            for (int r = 0; r < 8; ++r)
                k_ws[(size_t)(p0 + mbase + r) * 8 + coll] = acc[r];
        }
    }
    {                                           // v
        v8f acc = {};
        acc = WMMA_F16(a0, load_b_frag(wv, 0, 0, lane), acc);
        acc = WMMA_F16(a1, load_b_frag(wv, 0, 1, lane), acc);
        if (coll < 8) {
#pragma unroll
            for (int r = 0; r < 8; ++r)
                v_ws[(size_t)(p0 + mbase + r) * 8 + coll] = acc[r];
        }
    }
    __syncthreads();

    // one global atomicAdd pass per block for the q partials
    for (int i = tid; i < 16 * 64; i += 256) {
        const int M = i >> 6, c = i & 63;
        atomicAdd(&q_acc[(size_t)(l0 + M) * 64 + c], qred[M][c]);
    }
}

// ---------------------------------------------------------------------------
// K2: global attention per l. scores (8 x 512), softmax over n, o = attn @ v.
// ---------------------------------------------------------------------------
__global__ void __launch_bounds__(256)
attn_kernel(const float* __restrict__ k_ws, const float* __restrict__ v_ws,
            const float* __restrict__ q_acc, float* __restrict__ o_ws) {
    __shared__ float qs[64];
    __shared__ float sc[8][N_SEQ];
    __shared__ float vv[N_SEQ][8];
    __shared__ float part[4][64];

    const int l = blockIdx.x;
    const int tid = threadIdx.x;
    if (tid < 64) qs[tid] = q_acc[(size_t)l * 64 + tid] * (1.0f / (float)N_SEQ);
    __syncthreads();

    const float scale = 0.35355339059327373f;   // 1/sqrt(DH)
    for (int n = tid; n < N_SEQ; n += 256) {
        const float* kp = k_ws + ((size_t)n * L_SEQ + l) * 8;
        const float* vp = v_ws + ((size_t)n * L_SEQ + l) * 8;
        float kv[8];
#pragma unroll
        for (int d = 0; d < 8; ++d) kv[d] = kp[d];
#pragma unroll
        for (int d = 0; d < 8; ++d) vv[n][d] = vp[d];
#pragma unroll
        for (int h = 0; h < 8; ++h) {
            float dot = 0.0f;
#pragma unroll
            for (int d = 0; d < 8; ++d) dot += qs[h * 8 + d] * kv[d];
            sc[h][n] = dot * scale;
        }
    }
    __syncthreads();

    // softmax: wave h owns row h
    {
        const int h = tid >> 5, lane = tid & 31;
        float m = -1e30f;
        for (int n2 = lane; n2 < N_SEQ; n2 += 32) m = fmaxf(m, sc[h][n2]);
#pragma unroll
        for (int off = 16; off > 0; off >>= 1) m = fmaxf(m, __shfl_xor(m, off, 32));
        float ssum = 0.0f;
        for (int n2 = lane; n2 < N_SEQ; n2 += 32) {
            const float e = __expf(sc[h][n2] - m);
            sc[h][n2] = e;
            ssum += e;
        }
#pragma unroll
        for (int off = 16; off > 0; off >>= 1) ssum += __shfl_xor(ssum, off, 32);
        const float inv = 1.0f / ssum;
        for (int n2 = lane; n2 < N_SEQ; n2 += 32) sc[h][n2] *= inv;
    }
    __syncthreads();

    // o[h][dh] = sum_n attn[h][n] * v[n][dh], 4 partial segments per output
    {
        const int out = tid & 63;
        const int oh = out >> 3, od = out & 7;
        const int seg = tid >> 6;
        float p = 0.0f;
        const int nb = seg * 128;
        for (int n2 = nb; n2 < nb + 128; ++n2) p += sc[oh][n2] * vv[n2][od];
        part[seg][out] = p;
    }
    __syncthreads();
    if (tid < 64)
        o_ws[(size_t)l * 64 + tid] = part[0][tid] + part[1][tid] + part[2][tid] + part[3][tid];
}

// ---------------------------------------------------------------------------
// K3: gate = sigmoid(x@Wg + bg); out = (gate * o[l]) @ Wout + bout.
// Weights staged to LDS once per block; A fragments straight from the f16
// activation buffer.
// ---------------------------------------------------------------------------
__global__ void __launch_bounds__(256)
gate_out_kernel(const unsigned short* __restrict__ xbf, const float* __restrict__ o_ws,
                const float* __restrict__ bg, const float* __restrict__ bout,
                const unsigned short* __restrict__ wpack, float* __restrict__ out) {
    __shared__ __attribute__((aligned(16))) unsigned short tiles[8][16 * 64];
    __shared__ __attribute__((aligned(16))) unsigned short wlds[WP_GO_U16];
    const int tid  = threadIdx.x;
    const int wave = tid >> 5;
    const int lane = tid & 31;
    const int p0 = blockIdx.x * 128 + wave * 16;
    const int l0 = p0 & (L_SEQ - 1);
    const int mbase = (lane >> 4) << 3;
    const int coll  = lane & 15;

    // stage Wg|Wo fragments into LDS (1024 x 16B)
    {
        const U128* src = (const U128*)(wpack + WP_G);
        U128* dst = (U128*)wlds;
        for (int i = tid; i < WP_GO_U16 / 8; i += 256) dst[i] = src[i];
    }
    const unsigned short* wg = wlds;
    const unsigned short* wo = wlds + 4096;

    // A fragments straight from the stored f16 activations (row-major)
    const unsigned short* xtile = xbf + (size_t)p0 * 64;
    const v16h a0 = load_a_frag(xtile, lane, 0);
    const v16h a1 = load_a_frag(xtile, lane, 1);
    __syncthreads();   // weights staged

#pragma unroll
    for (int t = 0; t < 4; ++t) {
        v8f acc = {};
        acc = WMMA_F16(a0, load_b_frag(wg, t, 0, lane), acc);
        acc = WMMA_F16(a1, load_b_frag(wg, t, 1, lane), acc);
        const int col = t * 16 + coll;
        const float b = bg[col];
#pragma unroll
        for (int r = 0; r < 8; ++r) {
            const int M = mbase + r;
            const float g  = 1.0f / (1.0f + __expf(-(acc[r] + b)));
            const float tv = g * o_ws[(size_t)(l0 + M) * 64 + col];
            tiles[wave][M * 64 + col] = f2h_bits(tv);
        }
    }
    __syncthreads();

    const v16h g0 = load_a_frag(tiles[wave], lane, 0);
    const v16h g1 = load_a_frag(tiles[wave], lane, 1);
#pragma unroll
    for (int t = 0; t < 4; ++t) {
        v8f acc = {};
        acc = WMMA_F16(g0, load_b_frag(wo, t, 0, lane), acc);
        acc = WMMA_F16(g1, load_b_frag(wo, t, 1, lane), acc);
        const int col = t * 16 + coll;
        const float b = bout[col];
#pragma unroll
        for (int r = 0; r < 8; ++r)
            out[(size_t)(p0 + mbase + r) * 64 + col] = acc[r] + b;
    }
}

// ---------------------------------------------------------------------------
extern "C" void kernel_launch(void* const* d_in, const int* in_sizes, int n_in,
                              void* d_out, int out_size, void* d_ws, size_t ws_size,
                              hipStream_t stream) {
    const float* msa  = (const float*)d_in[0];
    const float* ln_w = (const float*)d_in[1];
    const float* ln_b = (const float*)d_in[2];
    const float* Wq   = (const float*)d_in[3];
    const float* Wk   = (const float*)d_in[4];
    const float* Wv   = (const float*)d_in[5];
    const float* Wg   = (const float*)d_in[6];
    const float* bg   = (const float*)d_in[7];
    const float* Wout = (const float*)d_in[8];
    const float* bout = (const float*)d_in[9];
    float* out = (float*)d_out;

    char* ws = (char*)d_ws;
    unsigned short* xbf   = (unsigned short*)(ws + OFF_XBF);
    float*          k_ws  = (float*)(ws + OFF_K);
    float*          v_ws  = (float*)(ws + OFF_V);
    float*          q_acc = (float*)(ws + OFF_Q);
    float*          o_ws  = (float*)(ws + OFF_O);
    unsigned short* wpack = (unsigned short*)(ws + OFF_WP);

    prepack_weights<<<1, 256, 0, stream>>>(Wq, Wk, Wv, Wg, Wout, wpack);
    hipMemsetAsync(q_acc, 0, (size_t)L_SEQ * HD * sizeof(float), stream);
    ln_qkv_kernel<<<4096, 256, 0, stream>>>(msa, ln_w, ln_b, xbf, k_ws, v_ws, q_acc, wpack);
    attn_kernel<<<L_SEQ, 256, 0, stream>>>(k_ws, v_ws, q_acc, o_ws);
    gate_out_kernel<<<4096, 256, 0, stream>>>(xbf, o_ws, bg, bout, wpack, out);
}
